// MultiheadAttention_39530878992772
// MI455X (gfx1250) — matrix operand order repaired and checked by reading, not compiled, not run on hardware
//
#include <hip/hip_runtime.h>
#include <math.h>

// ---------------------------------------------------------------------------
// MI455X (gfx1250) multihead attention, bf16 WMMA pipeline.
//   b=4, s=2048, E=1024, H=16, d_k=64.
// ---------------------------------------------------------------------------

typedef __attribute__((ext_vector_type(16))) __bf16 v16bf;
typedef __attribute__((ext_vector_type(8)))  float  v8f;

union FragB16 {
    v16bf v;
    uint4 q[2];
};

__device__ __forceinline__ unsigned short f2bf(float f) {
    unsigned int u = __float_as_uint(f);
    unsigned int r = u + 0x7FFFu + ((u >> 16) & 1u);   // round-to-nearest-even
    return (unsigned short)(r >> 16);
}

// A fragment (16x32 bf16, M in lanes, K in VGPRs) from row-major [*, ld].
// lane<16: row=lane, K = {k..k+7} in v0..3, {k+16..k+23} in v4..7
// lane>=16: row=lane-16, K = {k+8..k+15}, {k+24..k+31}
__device__ __forceinline__ void load_fragA(FragB16& f, const unsigned short* base,
                                           int row, int ld, int kBase, int half) {
    const unsigned short* p = base + (size_t)row * ld + kBase + half * 8;
    f.q[0] = *(const uint4*)(p);
    f.q[1] = *(const uint4*)(p + 16);
}

// B fragment (32x16 bf16, N in lanes, K in VGPRs) from a row-major TRANSPOSED
// array [N, Kdim] (i.e. per-lane column of B is a contiguous row here).
// lane<16: n=lane, K={kBase..+7},{kBase+16..+23} ; lane>=16: n=lane-16, K={kBase+8..+15},{kBase+24..+31}
__device__ __forceinline__ void load_fragB(FragB16& f, const unsigned short* baseT,
                                           int col, int ldT, int kBase, int half) {
    const unsigned short* p = baseT + (size_t)col * ldT + kBase + half * 8;
    f.q[0] = *(const uint4*)(p);
    f.q[1] = *(const uint4*)(p + 16);
}

__device__ __forceinline__ v8f wmma_bf16(const FragB16& a, const FragB16& b, v8f c) {
    return __builtin_amdgcn_wmma_f32_16x16x32_bf16(
        false, a.v, false, b.v, (short)0, c, false, false);
}

// ---------------------------------------------------------------------------
// conversion kernels
// ---------------------------------------------------------------------------
__global__ void cvt_f32_bf16(const float* __restrict__ in,
                             unsigned short* __restrict__ out, int n) {
    int i = blockIdx.x * blockDim.x + threadIdx.x;
    if (i < n) out[i] = f2bf(in[i]);
}

// W [K=1024, N=1024] row-major fp32 -> Wt [N, K] bf16
__global__ void cvt_transpose(const float* __restrict__ in,
                              unsigned short* __restrict__ outT, int Kd, int Nd) {
    int i = blockIdx.x * blockDim.x + threadIdx.x;
    if (i < Kd * Nd) {
        int k = i / Nd, n = i % Nd;
        outT[(size_t)n * Kd + k] = f2bf(in[i]);
    }
}

// ---------------------------------------------------------------------------
// QKV projection: out = x @ W + b.
//   Q, K stored head-major [b,h,s,d] bf16 (rows are WMMA A/B fragments).
//   V stored head-major TRANSPOSED [b,h,d,s] bf16 so the attention P@V
//   B-fragments are contiguous per lane straight from global (no LDS bounce).
// grid = (M/128, N/128, 3); block = 256 (8 waves); wave tile 32x64.
// Q additionally scaled by 1/sqrt(d_k)=0.125 (folded here).
// Weights are L2-resident (2MB each in 192MB L2) -> fragments straight from
// global memory, no LDS staging needed.
// ---------------------------------------------------------------------------
__global__ __launch_bounds__(256) void qkv_gemm(
    const unsigned short* __restrict__ xb,
    const unsigned short* __restrict__ wtq, const unsigned short* __restrict__ wtk,
    const unsigned short* __restrict__ wtv,
    const float* __restrict__ bq, const float* __restrict__ bk,
    const float* __restrict__ bv,
    unsigned short* __restrict__ Qh, unsigned short* __restrict__ Kh,
    unsigned short* __restrict__ Vt) {

    const int z = blockIdx.z;
    const unsigned short* Wt = (z == 0) ? wtq : (z == 1) ? wtk : wtv;
    const float* bias        = (z == 0) ? bq  : (z == 1) ? bk  : bv;
    unsigned short* out      = (z == 0) ? Qh  : (z == 1) ? Kh  : Vt;
    const float scale = (z == 0) ? 0.125f : 1.0f;

    const int tid = threadIdx.x;
    const int wave = tid >> 5, lane = tid & 31;
    const int half = lane >> 4, lr = lane & 15;
    const int wm = wave & 3, wn = wave >> 2;
    const int mBase = blockIdx.x * 128 + wm * 32;
    const int nBase = blockIdx.y * 128 + wn * 64;

    v8f acc[2][4];
#pragma unroll
    for (int mi = 0; mi < 2; ++mi)
#pragma unroll
        for (int ni = 0; ni < 4; ++ni) acc[mi][ni] = (v8f){};

    for (int k = 0; k < 1024; k += 32) {
        FragB16 a[2], b[4];
#pragma unroll
        for (int mi = 0; mi < 2; ++mi)
            load_fragA(a[mi], xb, mBase + mi * 16 + lr, 1024, k, half);
#pragma unroll
        for (int ni = 0; ni < 4; ++ni)
            load_fragB(b[ni], Wt, nBase + ni * 16 + lr, 1024, k, half);
        if (k + 128 < 1024) {   // stream-ahead: lowers to global_prefetch_b8
            __builtin_prefetch(xb + (size_t)(mBase + lr) * 1024 + k + 128, 0, 3);
            __builtin_prefetch(Wt + (size_t)(nBase + lr) * 1024 + k + 128, 0, 3);
        }
#pragma unroll
        for (int mi = 0; mi < 2; ++mi)
#pragma unroll
            for (int ni = 0; ni < 4; ++ni)
                acc[mi][ni] = wmma_bf16(a[mi], b[ni], acc[mi][ni]);
    }

    // epilogue: C layout lane (VGPR j, half) -> row j+8*half, col lr
#pragma unroll
    for (int ni = 0; ni < 4; ++ni) {
        const int n = nBase + ni * 16 + lr;
        const float bval = bias[n];
        const int h = n >> 6, d = n & 63;
#pragma unroll
        for (int mi = 0; mi < 2; ++mi) {
#pragma unroll
            for (int j = 0; j < 8; ++j) {
                const int m = mBase + mi * 16 + j + 8 * half;
                const int bb = m >> 11, s = m & 2047;
                const float v = (acc[mi][ni][j] + bval) * scale;
                if (z == 2)   // V transposed per head: [b,h,d,s]
                    out[(((size_t)(bb * 16 + h) * 64) + d) * 2048 + s] = f2bf(v);
                else          // Q/K head-major: [b,h,s,d]
                    out[(((size_t)(bb * 16 + h) * 2048) + s) * 64 + d] = f2bf(v);
            }
        }
    }
}

// ---------------------------------------------------------------------------
// Flash attention: grid = (2048/64, b*h=64), block = 128 (4 waves).
// Each wave owns 16 query rows; keys processed in chunks of 64.
//   S = Q K^T : Q A-fragments hoisted (loop-invariant); K rows are directly
//               the documented B layout (contiguous per lane, L2-resident).
//   online softmax via shfl_xor reductions in the 16-lane halves of C layout.
//   P bounced through LDS (C layout -> A layout), the only LDS use.
//   O += P V  : V^T rows from global are directly the B layout.
// Output merged [b,s,e] bf16.
// ---------------------------------------------------------------------------
__global__ __launch_bounds__(128) void attn_kernel(
    const unsigned short* __restrict__ Qh, const unsigned short* __restrict__ Kh,
    const unsigned short* __restrict__ Vt, unsigned short* __restrict__ merged) {

    __shared__ unsigned short Plds[4 * 16 * 64];   // per-wave 16x64 P, 8 KB

    const int bh = blockIdx.y;
    const int tid = threadIdx.x;
    const int wave = tid >> 5, lane = tid & 31;
    const int half = lane >> 4, lr = lane & 15;

    const unsigned short* Q  = Qh + (size_t)bh * 2048 * 64;
    const unsigned short* Kp = Kh + (size_t)bh * 2048 * 64;
    const unsigned short* Vp = Vt + (size_t)bh * 64 * 2048;   // [d][s]

    const int qRow0 = blockIdx.x * 64 + wave * 16;
    unsigned short* Pw = Plds + wave * 1024;

    FragB16 qa[2];                                  // loop-invariant Q frags
    load_fragA(qa[0], Q, qRow0 + lr, 64, 0, half);
    load_fragA(qa[1], Q, qRow0 + lr, 64, 32, half);

    v8f o[4];
#pragma unroll
    for (int dt = 0; dt < 4; ++dt) o[dt] = (v8f){};
    float mrow[8], lsum[8];
#pragma unroll
    for (int j = 0; j < 8; ++j) { mrow[j] = -INFINITY; lsum[j] = 0.0f; }

    for (int kc = 0; kc < 2048; kc += 64) {
        // ---- S = Q K^T  (16 x 64 scores: four 16-col C tiles, 2 k-steps) ---
        v8f s[4];
#pragma unroll
        for (int nt = 0; nt < 4; ++nt) s[nt] = (v8f){};
#pragma unroll
        for (int nt = 0; nt < 4; ++nt) {
#pragma unroll
            for (int kk = 0; kk < 2; ++kk) {
                FragB16 b;
                load_fragB(b, Kp, kc + nt * 16 + lr, 64, kk * 32, half);
                s[nt] = wmma_bf16(qa[kk], b, s[nt]);
            }
        }

        // ---- online softmax (rows live in VGPR index j, half selects +8) ---
        float scl[8];
#pragma unroll
        for (int j = 0; j < 8; ++j) {
            float v = fmaxf(fmaxf(s[0][j], s[1][j]), fmaxf(s[2][j], s[3][j]));
#pragma unroll
            for (int mk = 1; mk <= 8; mk <<= 1) v = fmaxf(v, __shfl_xor(v, mk, 32));
            const float mn = fmaxf(mrow[j], v);
            scl[j] = __expf(mrow[j] - mn);
            mrow[j] = mn;
        }
#pragma unroll
        for (int nt = 0; nt < 4; ++nt)
#pragma unroll
            for (int j = 0; j < 8; ++j) s[nt][j] = __expf(s[nt][j] - mrow[j]);
#pragma unroll
        for (int j = 0; j < 8; ++j) {
            float r = (s[0][j] + s[1][j]) + (s[2][j] + s[3][j]);
#pragma unroll
            for (int mk = 1; mk <= 8; mk <<= 1) r += __shfl_xor(r, mk, 32);
            lsum[j] = lsum[j] * scl[j] + r;
#pragma unroll
            for (int dt = 0; dt < 4; ++dt) o[dt][j] *= scl[j];
        }

        // ---- bounce P through LDS: C layout -> A layout ----
#pragma unroll
        for (int nt = 0; nt < 4; ++nt)
#pragma unroll
            for (int j = 0; j < 8; ++j)
                Pw[(j + 8 * half) * 64 + nt * 16 + lr] = f2bf(s[nt][j]);
        __syncthreads();

        // ---- O += P @ V  (2 k-steps of 32 keys, 4 d-tiles) ----
        FragB16 pa[2];
        load_fragA(pa[0], Pw, lr, 64, 0, half);
        load_fragA(pa[1], Pw, lr, 64, 32, half);
#pragma unroll
        for (int dt = 0; dt < 4; ++dt) {
#pragma unroll
            for (int kk = 0; kk < 2; ++kk) {
                FragB16 vb;
                load_fragB(vb, Vp, dt * 16 + lr, 2048, kc + kk * 32, half);
                o[dt] = wmma_bf16(pa[kk], vb, o[dt]);
            }
        }
        __syncthreads();   // WAR: P reads done before next chunk's writes
    }

    // ---- finalize: O /= l, store merged [b, s, h*64+d] ----
    const int bb = bh >> 4, h = bh & 15;
#pragma unroll
    for (int j = 0; j < 8; ++j) {
        const float inv = 1.0f / lsum[j];
        const int srow = qRow0 + j + 8 * half;
#pragma unroll
        for (int dt = 0; dt < 4; ++dt)
            merged[((size_t)(bb * 2048 + srow)) * 1024 + h * 64 + dt * 16 + lr] =
                f2bf(o[dt][j] * inv);
    }
}

// ---------------------------------------------------------------------------
// output projection: out(fp32) = merged @ w_o + b_o
// grid = (64, 8), block 256, same tiling as qkv_gemm.
// ---------------------------------------------------------------------------
__global__ __launch_bounds__(256) void oproj_gemm(
    const unsigned short* __restrict__ A, const unsigned short* __restrict__ Wt,
    const float* __restrict__ bias, float* __restrict__ out) {

    const int tid = threadIdx.x;
    const int wave = tid >> 5, lane = tid & 31;
    const int half = lane >> 4, lr = lane & 15;
    const int wm = wave & 3, wn = wave >> 2;
    const int mBase = blockIdx.x * 128 + wm * 32;
    const int nBase = blockIdx.y * 128 + wn * 64;

    v8f acc[2][4];
#pragma unroll
    for (int mi = 0; mi < 2; ++mi)
#pragma unroll
        for (int ni = 0; ni < 4; ++ni) acc[mi][ni] = (v8f){};

    for (int k = 0; k < 1024; k += 32) {
        FragB16 a[2], b[4];
#pragma unroll
        for (int mi = 0; mi < 2; ++mi)
            load_fragA(a[mi], A, mBase + mi * 16 + lr, 1024, k, half);
#pragma unroll
        for (int ni = 0; ni < 4; ++ni)
            load_fragB(b[ni], Wt, nBase + ni * 16 + lr, 1024, k, half);
        if (k + 128 < 1024) {   // stream-ahead: lowers to global_prefetch_b8
            __builtin_prefetch(A  + (size_t)(mBase + lr) * 1024 + k + 128, 0, 3);
            __builtin_prefetch(Wt + (size_t)(nBase + lr) * 1024 + k + 128, 0, 3);
        }
#pragma unroll
        for (int mi = 0; mi < 2; ++mi)
#pragma unroll
            for (int ni = 0; ni < 4; ++ni)
                acc[mi][ni] = wmma_bf16(a[mi], b[ni], acc[mi][ni]);
    }

#pragma unroll
    for (int ni = 0; ni < 4; ++ni) {
        const int n = nBase + ni * 16 + lr;
        const float bval = bias[n];
#pragma unroll
        for (int mi = 0; mi < 2; ++mi)
#pragma unroll
            for (int j = 0; j < 8; ++j) {
                const int m = mBase + mi * 16 + j + 8 * half;
                out[(size_t)m * 1024 + n] = acc[mi][ni][j] + bval;
            }
    }
}

// ---------------------------------------------------------------------------
// launch
// ---------------------------------------------------------------------------
extern "C" void kernel_launch(void* const* d_in, const int* in_sizes, int n_in,
                              void* d_out, int out_size, void* d_ws, size_t ws_size,
                              hipStream_t stream) {
    const float* x   = (const float*)d_in[0];
    const float* w_q = (const float*)d_in[1];
    const float* b_q = (const float*)d_in[2];
    const float* w_k = (const float*)d_in[3];
    const float* b_k = (const float*)d_in[4];
    const float* w_v = (const float*)d_in[5];
    const float* b_v = (const float*)d_in[6];
    const float* w_o = (const float*)d_in[7];
    const float* b_o = (const float*)d_in[8];
    float* out = (float*)d_out;

    // workspace carve-up (bf16 everywhere): 88 MiB total
    char* ws = (char*)d_ws;
    const size_t N_X = (size_t)8192 * 1024;   // x / merged elements
    const size_t N_W = (size_t)1024 * 1024;   // weight elements
    unsigned short* xb     = (unsigned short*)ws;                    ws += N_X * 2;
    unsigned short* wtq    = (unsigned short*)ws;                    ws += N_W * 2;
    unsigned short* wtk    = (unsigned short*)ws;                    ws += N_W * 2;
    unsigned short* wtv    = (unsigned short*)ws;                    ws += N_W * 2;
    unsigned short* wto    = (unsigned short*)ws;                    ws += N_W * 2;
    unsigned short* Qh     = (unsigned short*)ws;                    ws += N_X * 2;
    unsigned short* Kh     = (unsigned short*)ws;                    ws += N_X * 2;
    unsigned short* Vt     = (unsigned short*)ws;                    ws += N_X * 2;
    unsigned short* merged = (unsigned short*)ws;

    // 1) convert x, transpose+convert weights
    cvt_f32_bf16<<<dim3((N_X + 255) / 256), dim3(256), 0, stream>>>(x, xb, (int)N_X);
    cvt_transpose<<<dim3((N_W + 255) / 256), dim3(256), 0, stream>>>(w_q, wtq, 1024, 1024);
    cvt_transpose<<<dim3((N_W + 255) / 256), dim3(256), 0, stream>>>(w_k, wtk, 1024, 1024);
    cvt_transpose<<<dim3((N_W + 255) / 256), dim3(256), 0, stream>>>(w_v, wtv, 1024, 1024);
    cvt_transpose<<<dim3((N_W + 255) / 256), dim3(256), 0, stream>>>(w_o, wto, 1024, 1024);

    // 2) fused QKV projection (z selects Q/K/V); Q pre-scaled by 1/sqrt(64);
    //    V written transposed per head.
    qkv_gemm<<<dim3(64, 8, 3), dim3(256), 0, stream>>>(
        xb, wtq, wtk, wtv, b_q, b_k, b_v, Qh, Kh, Vt);

    // 3) flash attention, merged [b,s,e] bf16
    attn_kernel<<<dim3(32, 64), dim3(128), 0, stream>>>(Qh, Kh, Vt, merged);

    // 4) output projection (fp32 + bias)
    oproj_gemm<<<dim3(64, 8), dim3(256), 0, stream>>>(merged, wto, b_o, out);
}
